// MutualTimeSformer_1425929142991
// MI455X (gfx1250) — matrix operand order repaired and checked by reading, not compiled, not run on hardware
//
#include <hip/hip_runtime.h>
#include <hip/hip_bf16.h>

// ---------------------------------------------------------------------------
// MutualTimeSformer forward for MI455X (gfx1250, wave32, WMMA).
// All dense GEMMs run on v_wmma_f32_16x16x32_bf16 (bf16 A/B, fp32 accum).
// Each wave computes a 32x64 output tile: 2 A-frags x 4 B-frags = 8 WMMAs
// per 32-K step against 12 b128 loads, with all fragment loads batched
// ahead of the WMMA chain for intra-wave latency hiding. Weights pre-packed
// to B-fragment layout once per launch. Attention runs on VALU fp32.
// ---------------------------------------------------------------------------

typedef unsigned short u16;
typedef unsigned int   u32;

typedef __attribute__((ext_vector_type(16))) __bf16 v16bf;
typedef __attribute__((ext_vector_type(8)))  float  v8f;

union Frag32B { uint4 u[2]; v16bf v; };

// Model constants
#define D_MODEL   512
#define D_QKV     1536
#define D_FF1     4096
#define D_FF2     2048
#define PD        768
#define NTOK      4800
#define NSEQ      4801
#define MPAD      4864      // 19 blocks * 256 rows: grid tiles M exactly
#define NHEAD     8
#define DH        64
#define DEPTH     12
#define QSCALE    0.125f    // 64^-0.5

__device__ __forceinline__ u16 f2bf(float f) {
    union { float f; u32 u; } c; c.f = f;
    u32 u = c.u + 0x7FFFu + ((c.u >> 16) & 1u);
    return (u16)(u >> 16);
}

// ---------------------------------------------------------------------------
// Weight packing: fp32 W[layer][K][N] -> bf16 WMMA-B fragment layout.
// Block (nt,kt) covers B[kt*32 .. +32)[nt*16 .. +16); 512 elems contiguous:
//   elem index = lane*16 + e ; value = W[kt*32 + 8*(lane/16) + (e%8) + 16*(e/8)][nt*16 + lane%16]
// ---------------------------------------------------------------------------
__global__ __launch_bounds__(256) void pack_b(const float* __restrict__ W,
                                              u16* __restrict__ P,
                                              int K, int N, int layers) {
    size_t tid = (size_t)blockIdx.x * 256 + threadIdx.x;
    size_t per = (size_t)K * N;
    if (tid >= per * layers) return;
    int    layer = (int)(tid / per);
    size_t r     = tid % per;
    int Kt   = K >> 5;
    size_t blk = r >> 9;           // 512 elems per 16x32 block
    int e    = (int)(r & 15);
    int lane = (int)((r >> 4) & 31);
    int nt   = (int)(blk / Kt);
    int kt   = (int)(blk % Kt);
    int k    = kt * 32 + ((lane >> 4) << 3) + (e & 7) + ((e >> 3) << 4);
    int n    = nt * 16 + (lane & 15);
    P[tid] = f2bf(W[layer * per + (size_t)k * N + n]);
}

// ---------------------------------------------------------------------------
// Tiled bf16 WMMA GEMM: C[m,n] = sum_k A[m,k] * B[k,n]  (+ fused epilogue)
//   A: bf16 row-major [MPAD x lda], B: packed fragments, out: fp32.
//   Wave tile 32x64; block = 8 waves = 256 rows; grid.y = MPAD/256 = 19.
// MODE 0: patch embed  : out[(m+1)*ldo+n] = acc + bias[n] + aux[(m+1)*ldo+n], m<4800
// MODE 1: qkv          : out[m*ldo+n] = acc * (n<512 ? QSCALE : 1)
// MODE 2: residual+bias: out[m*ldo+n] += acc + bias[n]
// MODE 3: bias         : out[m*ldo+n] = acc + bias[n]
// ---------------------------------------------------------------------------
template <int MODE>
__global__ __launch_bounds__(256) void gemm_bf16(const u16* __restrict__ A, int lda,
                                                 const u16* __restrict__ Bp, int K,
                                                 const float* __restrict__ bias,
                                                 const float* __restrict__ aux,
                                                 float* __restrict__ out, int ldo) {
    const int lane = threadIdx.x & 31;
    const int wave = threadIdx.x >> 5;
    const int ng   = blockIdx.x;                         // group of 64 columns
    const int Kt   = K >> 5;
    const int m0   = (blockIdx.y * 8 + wave) << 5;       // 32 rows per wave
    const int half = lane >> 4;
    const int lm   = lane & 15;

    const size_t rowq = (size_t)(lda >> 3);              // uint4 per A row
    const uint4* Aq0 = reinterpret_cast<const uint4*>(A) + (size_t)(m0 + lm) * rowq;
    const uint4* Aq1 = Aq0 + 16 * rowq;
    const uint4* Bq  = reinterpret_cast<const uint4*>(Bp)
                     + (size_t)(ng * 4) * Kt * 64 + lane * 2;

    v8f acc[2][4] = {};

    for (int kt = 0; kt < Kt; ++kt) {
        // Batch all 12 b128 loads of this K-step into distinct registers so
        // the compiler can clause them and wait incrementally, overlapping
        // the WMMA chain with outstanding loads.
        Frag32B a0, a1, bf[4];
        a0.u[0] = Aq0[kt * 4 + half];
        a0.u[1] = Aq0[kt * 4 + 2 + half];
        a1.u[0] = Aq1[kt * 4 + half];
        a1.u[1] = Aq1[kt * 4 + 2 + half];
#pragma unroll
        for (int s = 0; s < 4; ++s) {
            const uint4* bq = Bq + ((size_t)s * Kt + kt) * 64;
            bf[s].u[0] = bq[0];
            bf[s].u[1] = bq[1];
        }
#pragma unroll
        for (int s = 0; s < 4; ++s) {
            acc[0][s] = __builtin_amdgcn_wmma_f32_16x16x32_bf16(
                false, a0.v, false, bf[s].v, (short)0, acc[0][s], false, false);
            acc[1][s] = __builtin_amdgcn_wmma_f32_16x16x32_bf16(
                false, a1.v, false, bf[s].v, (short)0, acc[1][s], false, false);
        }
    }

#pragma unroll
    for (int mi = 0; mi < 2; ++mi) {
#pragma unroll
        for (int s = 0; s < 4; ++s) {
            const int n = ng * 64 + s * 16 + lm;
            const float bv = (MODE == 0 || MODE == 2 || MODE == 3) ? bias[n] : 0.f;
#pragma unroll
            for (int r = 0; r < 8; ++r) {
                const int m = m0 + mi * 16 + r + half * 8;
                float v = acc[mi][s][r];
                if (MODE == 0) {
                    if (m < NTOK) {
                        size_t o = (size_t)(m + 1) * ldo + n;
                        out[o] = v + bv + aux[o];
                    }
                } else if (MODE == 1) {
                    out[(size_t)m * ldo + n] = (n < D_MODEL) ? v * QSCALE : v;
                } else if (MODE == 2) {
                    size_t o = (size_t)m * ldo + n;
                    out[o] += v + bv;
                } else {
                    out[(size_t)m * ldo + n] = v + bv;
                }
            }
        }
    }
}

// ---------------------------------------------------------------------------
// Patchify: video [1,4,3,480,640] -> tok bf16 [NTOK x 768]
// token t = g*2400 + f*600 + hi*20 + wi ; elem p = (p1*16+p2)*3 + c
// ---------------------------------------------------------------------------
__global__ __launch_bounds__(256) void patchify_k(const float* __restrict__ video,
                                                  u16* __restrict__ tok) {
    size_t tid = (size_t)blockIdx.x * 256 + threadIdx.x;
    if (tid >= (size_t)NTOK * PD) return;
    int t = (int)(tid / PD), p = (int)(tid % PD);
    int g  = t / 2400, u = t % 2400;
    int f  = u / 600;  int hw = u % 600;
    int hi = hw / 20,  wi = hw % 20;
    int pp = p / 3,    c  = p % 3;
    int p1 = pp >> 4,  p2 = pp & 15;
    int row = hi * 16 + p1;
    int col = g * 320 + wi * 16 + p2;
    float v = video[(((size_t)f * 3 + c) * 480 + row) * 640 + col];
    tok[tid] = f2bf(v);
}

__global__ void cls_init_k(const float* __restrict__ cls_tok,
                           const float* __restrict__ pos,
                           float* __restrict__ x) {
    int d = blockIdx.x * 256 + threadIdx.x;
    if (d < D_MODEL) x[d] = cls_tok[d] + pos[d];
}

// ---------------------------------------------------------------------------
// Row layernorm: x fp32 [nrows x 512] -> xn bf16. One wave per row.
// ---------------------------------------------------------------------------
__global__ __launch_bounds__(256) void ln_rows(const float* __restrict__ x,
                                               const float* __restrict__ g,
                                               const float* __restrict__ b,
                                               u16* __restrict__ xn, int nrows) {
    int wave = threadIdx.x >> 5, lane = threadIdx.x & 31;
    int row = blockIdx.x * 8 + wave;
    if (row >= nrows) return;
    const float4* xr = reinterpret_cast<const float4*>(x + (size_t)row * D_MODEL) + lane * 4;
    float4 d0 = xr[0], d1 = xr[1], d2 = xr[2], d3 = xr[3];
    float vals[16] = {d0.x,d0.y,d0.z,d0.w, d1.x,d1.y,d1.z,d1.w,
                      d2.x,d2.y,d2.z,d2.w, d3.x,d3.y,d3.z,d3.w};
    float s = 0.f, sq = 0.f;
#pragma unroll
    for (int k = 0; k < 16; ++k) { s += vals[k]; sq += vals[k] * vals[k]; }
#pragma unroll
    for (int o = 16; o >= 1; o >>= 1) {
        s  += __shfl_xor(s,  o, 32);
        sq += __shfl_xor(sq, o, 32);
    }
    float mu = s * (1.f / D_MODEL);
    float var = sq * (1.f / D_MODEL) - mu * mu;
    float rs = rsqrtf(var + 1e-5f);
    int base = lane * 16;
    u32 w[8];
#pragma unroll
    for (int k = 0; k < 8; ++k) {
        u16 lo = f2bf((vals[2*k]   - mu) * rs * g[base + 2*k]   + b[base + 2*k]);
        u16 hi = f2bf((vals[2*k+1] - mu) * rs * g[base + 2*k+1] + b[base + 2*k+1]);
        w[k] = (u32)lo | ((u32)hi << 16);
    }
    uint4* xo = reinterpret_cast<uint4*>(xn + (size_t)row * D_MODEL) + lane * 2;
    xo[0] = make_uint4(w[0], w[1], w[2], w[3]);
    xo[1] = make_uint4(w[4], w[5], w[6], w[7]);
}

// ---------------------------------------------------------------------------
// cls attention: per head, q = row 0, keys/values = rows 0..4800.
// ---------------------------------------------------------------------------
__global__ __launch_bounds__(256) void cls_attn(const float* __restrict__ qkv,
                                                u16* __restrict__ attn) {
    __shared__ float sc[NSEQ];
    __shared__ float red[256];
    const int hd = blockIdx.x, tid = threadIdx.x;
    const float* qr = qkv + hd * DH;     // row 0 Q (already scaled)
    float lmax = -1e30f;
    for (int j = tid; j < NSEQ; j += 256) {
        const float* kr = qkv + (size_t)j * D_QKV + D_MODEL + hd * DH;
        float s = 0.f;
#pragma unroll
        for (int d = 0; d < DH; ++d) s += qr[d] * kr[d];
        sc[j] = s;
        lmax = fmaxf(lmax, s);
    }
    red[tid] = lmax; __syncthreads();
    for (int o = 128; o >= 1; o >>= 1) {
        if (tid < o) red[tid] = fmaxf(red[tid], red[tid + o]);
        __syncthreads();
    }
    float mx = red[0]; __syncthreads();
    float ls = 0.f;
    for (int j = tid; j < NSEQ; j += 256) {
        float e = __expf(sc[j] - mx);
        sc[j] = e; ls += e;
    }
    red[tid] = ls; __syncthreads();
    for (int o = 128; o >= 1; o >>= 1) {
        if (tid < o) red[tid] += red[tid + o];
        __syncthreads();
    }
    float inv = 1.f / red[0]; __syncthreads();
    int d = tid & 63, part = tid >> 6;
    float acc = 0.f;
    for (int j = part; j < NSEQ; j += 4)
        acc += sc[j] * qkv[(size_t)j * D_QKV + 1024 + hd * DH + d];
    red[tid] = acc; __syncthreads();
    if (tid < 64) {
        float o = (red[tid] + red[tid + 64] + red[tid + 128] + red[tid + 192]) * inv;
        attn[hd * DH + tid] = f2bf(o);   // row 0
    }
}

// ---------------------------------------------------------------------------
// time attention: query token t attends [cls, same n_idx across 4 frames].
// ---------------------------------------------------------------------------
__global__ __launch_bounds__(256) void time_attn(const float* __restrict__ qkv,
                                                 u16* __restrict__ attn) {
    int gid = blockIdx.x * 256 + threadIdx.x;
    if (gid >= NHEAD * NTOK) return;
    int t = gid % NTOK, hd = gid / NTOK;
    int n_idx = t % 1200;
    const float* qr = qkv + (size_t)(1 + t) * D_QKV + hd * DH;
    const float* kb[5];
    const float* vb[5];
#pragma unroll
    for (int j = 0; j < 5; ++j) {
        int row = (j == 0) ? 0 : 1 + (j - 1) * 1200 + n_idx;
        kb[j] = qkv + (size_t)row * D_QKV + D_MODEL + hd * DH;
        vb[j] = kb[j] + D_MODEL;
    }
    float s[5] = {0.f, 0.f, 0.f, 0.f, 0.f};
    for (int d = 0; d < DH; ++d) {
        float qd = qr[d];
#pragma unroll
        for (int j = 0; j < 5; ++j) s[j] += qd * kb[j][d];
    }
    float mx = s[0];
#pragma unroll
    for (int j = 1; j < 5; ++j) mx = fmaxf(mx, s[j]);
    float e[5], l = 0.f;
#pragma unroll
    for (int j = 0; j < 5; ++j) { e[j] = __expf(s[j] - mx); l += e[j]; }
    float inv = 1.f / l;
    u16* orow = attn + (size_t)(1 + t) * D_MODEL + hd * DH;
    for (int d = 0; d < DH; ++d) {
        float o = 0.f;
#pragma unroll
        for (int j = 0; j < 5; ++j) o += e[j] * vb[j][d];
        orow[d] = f2bf(o * inv);
    }
}

// ---------------------------------------------------------------------------
// space attention (mutual halves): block = (head, frame, half). 600 queries
// over 601 keys [cls + other half]. All threads walk keys in lockstep so the
// key/value rows are uniform-address broadcast loads.
// ---------------------------------------------------------------------------
__global__ __launch_bounds__(256) void space_attn(const float* __restrict__ qkv,
                                                  u16* __restrict__ attn) {
    int bx = blockIdx.x;
    int hd = bx & 7, f = (bx >> 3) & 3, sel = bx >> 5;
    for (int i = threadIdx.x; i < 600; i += 256) {
        int tq = f * 1200 + sel * 600 + i;
        const float* qr = qkv + (size_t)(1 + tq) * D_QKV + hd * DH;
        float q[DH];
#pragma unroll
        for (int d = 0; d < DH; ++d) q[d] = qr[d];
        float m = -1e30f, l = 0.f;
        float acc[DH];
#pragma unroll
        for (int d = 0; d < DH; ++d) acc[d] = 0.f;
        for (int j = 0; j < 601; ++j) {
            int row = (j == 0) ? 0 : 1 + f * 1200 + (1 - sel) * 600 + (j - 1);
            const float* kr = qkv + (size_t)row * D_QKV + D_MODEL + hd * DH;
            const float* vr = kr + D_MODEL;
            __builtin_prefetch((const void*)(kr + D_QKV), 0, 0);
            float s = 0.f;
#pragma unroll
            for (int d = 0; d < DH; ++d) s += q[d] * kr[d];
            float mn = fmaxf(m, s);
            float c = __expf(m - mn), p = __expf(s - mn);
            l = l * c + p;
#pragma unroll
            for (int d = 0; d < DH; ++d) acc[d] = acc[d] * c + p * vr[d];
            m = mn;
        }
        float inv = 1.f / l;
        u16* orow = attn + (size_t)(1 + tq) * D_MODEL + hd * DH;
        for (int d = 0; d < DH; ++d) orow[d] = f2bf(acc[d] * inv);
    }
}

// ---------------------------------------------------------------------------
// GEGLU: h fp32 [MPAD x 4096] -> ug bf16 [MPAD x 2048], u * gelu_exact(g)
// ---------------------------------------------------------------------------
__global__ __launch_bounds__(256) void geglu_k(const float* __restrict__ h,
                                               u16* __restrict__ ug) {
    size_t tid = (size_t)blockIdx.x * 256 + threadIdx.x;
    if (tid >= (size_t)MPAD * D_FF2) return;
    size_t row = tid >> 11;
    int col = (int)(tid & 2047);
    float u = h[row * D_FF1 + col];
    float g = h[row * D_FF1 + D_FF2 + col];
    float gg = 0.5f * g * (1.f + erff(g * 0.70710678118654752f));
    ug[tid] = f2bf(u * gg);
}

// ---------------------------------------------------------------------------
// Head: layernorm(x[0]) @ o_w [512x60] + o_b
// ---------------------------------------------------------------------------
__global__ __launch_bounds__(256) void head_k(const float* __restrict__ x,
                                              const float* __restrict__ g,
                                              const float* __restrict__ b,
                                              const float* __restrict__ ow,
                                              const float* __restrict__ ob,
                                              float* __restrict__ out) {
    __shared__ float r1[256], r2[256], mv[2];
    int tid = threadIdx.x;
    float a = x[tid], c = x[tid + 256];
    r1[tid] = a + c; r2[tid] = a * a + c * c;
    __syncthreads();
    for (int o = 128; o >= 1; o >>= 1) {
        if (tid < o) { r1[tid] += r1[tid + o]; r2[tid] += r2[tid + o]; }
        __syncthreads();
    }
    if (tid == 0) {
        float mu = r1[0] * (1.f / D_MODEL);
        float var = r2[0] * (1.f / D_MODEL) - mu * mu;
        mv[0] = mu; mv[1] = rsqrtf(var + 1e-5f);
    }
    __syncthreads();
    float mu = mv[0], rs = mv[1];
    if (tid < 60) {
        float acc = 0.f;
        for (int d = 0; d < D_MODEL; ++d) {
            float xn = (x[d] - mu) * rs * g[d] + b[d];
            acc += xn * ow[(size_t)d * 60 + tid];
        }
        out[tid] = acc + ob[tid];
    }
}

// ---------------------------------------------------------------------------
// Host-side orchestration
// ---------------------------------------------------------------------------
static inline int cdiv(size_t a, size_t b) { return (int)((a + b - 1) / b); }

extern "C" void kernel_launch(void* const* d_in, const int* in_sizes, int n_in,
                              void* d_out, int out_size, void* d_ws, size_t ws_size,
                              hipStream_t stream) {
    const float* video   = (const float*)d_in[0];
    const float* patch_w = (const float*)d_in[1];
    const float* patch_b = (const float*)d_in[2];
    const float* pos_emb = (const float*)d_in[3];
    const float* cls_tok = (const float*)d_in[4];
    const float* t_ln_g  = (const float*)d_in[5];
    const float* t_ln_b  = (const float*)d_in[6];
    const float* t_qkv_w = (const float*)d_in[7];
    const float* t_out_w = (const float*)d_in[8];
    const float* t_out_b = (const float*)d_in[9];
    const float* s_ln_g  = (const float*)d_in[10];
    const float* s_ln_b  = (const float*)d_in[11];
    const float* s_qkv_w = (const float*)d_in[12];
    const float* s_out_w = (const float*)d_in[13];
    const float* s_out_b = (const float*)d_in[14];
    const float* f_ln_g  = (const float*)d_in[15];
    const float* f_ln_b  = (const float*)d_in[16];
    const float* f_w1    = (const float*)d_in[17];
    const float* f_b1    = (const float*)d_in[18];
    const float* f_w2    = (const float*)d_in[19];
    const float* f_b2    = (const float*)d_in[20];
    const float* o_ln_g  = (const float*)d_in[21];
    const float* o_ln_b  = (const float*)d_in[22];
    const float* o_w     = (const float*)d_in[23];
    const float* o_b     = (const float*)d_in[24];
    float* out = (float*)d_out;

    // ---- workspace carve (256B aligned) ----
    char* ws = (char*)d_ws;
    size_t off = 0;
    auto alloc = [&](size_t bytes) -> void* {
        off = (off + 255) & ~(size_t)255;
        void* p = ws + off;
        off += bytes;
        return p;
    };
    u16* wp_patch = (u16*)alloc((size_t)PD * D_MODEL * 2);
    u16* wp_tqkv  = (u16*)alloc((size_t)DEPTH * D_MODEL * D_QKV * 2);
    u16* wp_tout  = (u16*)alloc((size_t)DEPTH * D_MODEL * D_MODEL * 2);
    u16* wp_sqkv  = (u16*)alloc((size_t)DEPTH * D_MODEL * D_QKV * 2);
    u16* wp_sout  = (u16*)alloc((size_t)DEPTH * D_MODEL * D_MODEL * 2);
    u16* wp_fw1   = (u16*)alloc((size_t)DEPTH * D_MODEL * D_FF1 * 2);
    u16* wp_fw2   = (u16*)alloc((size_t)DEPTH * D_FF2 * D_MODEL * 2);
    u16*   tok  = (u16*)  alloc((size_t)MPAD * PD * 2);
    float* x    = (float*)alloc((size_t)MPAD * D_MODEL * 4);
    u16*   xn   = (u16*)  alloc((size_t)MPAD * D_MODEL * 2);
    float* qkv  = (float*)alloc((size_t)MPAD * D_QKV * 4);
    u16*   attn = (u16*)  alloc((size_t)MPAD * D_MODEL * 2);
    float* hbuf = (float*)alloc((size_t)MPAD * D_FF1 * 4);
    u16*   ug   = (u16*)  alloc((size_t)MPAD * D_FF2 * 2);
    if (off > ws_size) return;   // insufficient scratch: bail deterministically

    // ---- pack weights to bf16 WMMA fragment layout ----
    auto pack = [&](const float* W, u16* P, int K, int N, int L) {
        size_t total = (size_t)K * N * L;
        pack_b<<<cdiv(total, 256), 256, 0, stream>>>(W, P, K, N, L);
    };
    pack(patch_w, wp_patch, PD, D_MODEL, 1);
    pack(t_qkv_w, wp_tqkv, D_MODEL, D_QKV, DEPTH);
    pack(t_out_w, wp_tout, D_MODEL, D_MODEL, DEPTH);
    pack(s_qkv_w, wp_sqkv, D_MODEL, D_QKV, DEPTH);
    pack(s_out_w, wp_sout, D_MODEL, D_MODEL, DEPTH);
    pack(f_w1,    wp_fw1,  D_MODEL, D_FF1, DEPTH);
    pack(f_w2,    wp_fw2,  D_FF2,   D_MODEL, DEPTH);

    // ---- embed ----
    patchify_k<<<cdiv((size_t)NTOK * PD, 256), 256, 0, stream>>>(video, tok);
    cls_init_k<<<2, 256, 0, stream>>>(cls_tok, pos_emb, x);
    // x[1..4800] = tok @ patch_w + patch_b + pos_emb
    const int GY = MPAD / 256;              // 19: exact M coverage, no guards
    gemm_bf16<0><<<dim3(D_MODEL / 64, GY), 256, 0, stream>>>(
        tok, PD, wp_patch, PD, patch_b, pos_emb, x, D_MODEL);

    const int LN_GRID = MPAD / 8;           // 608
    const dim3 gQKV(D_QKV / 64, GY), gOUT(D_MODEL / 64, GY), gFF1(D_FF1 / 64, GY);

    for (int i = 0; i < DEPTH; ++i) {
        // ---- time attention block ----
        ln_rows<<<LN_GRID, 256, 0, stream>>>(x, t_ln_g + i * D_MODEL, t_ln_b + i * D_MODEL, xn, MPAD);
        gemm_bf16<1><<<gQKV, 256, 0, stream>>>(
            xn, D_MODEL, wp_tqkv + (size_t)i * D_MODEL * D_QKV, D_MODEL,
            nullptr, nullptr, qkv, D_QKV);
        cls_attn<<<NHEAD, 256, 0, stream>>>(qkv, attn);
        time_attn<<<cdiv((size_t)NHEAD * NTOK, 256), 256, 0, stream>>>(qkv, attn);
        gemm_bf16<2><<<gOUT, 256, 0, stream>>>(
            attn, D_MODEL, wp_tout + (size_t)i * D_MODEL * D_MODEL, D_MODEL,
            t_out_b + i * D_MODEL, nullptr, x, D_MODEL);

        // ---- space attention block ----
        ln_rows<<<LN_GRID, 256, 0, stream>>>(x, s_ln_g + i * D_MODEL, s_ln_b + i * D_MODEL, xn, MPAD);
        gemm_bf16<1><<<gQKV, 256, 0, stream>>>(
            xn, D_MODEL, wp_sqkv + (size_t)i * D_MODEL * D_QKV, D_MODEL,
            nullptr, nullptr, qkv, D_QKV);
        cls_attn<<<NHEAD, 256, 0, stream>>>(qkv, attn);
        space_attn<<<64, 256, 0, stream>>>(qkv, attn);
        gemm_bf16<2><<<gOUT, 256, 0, stream>>>(
            attn, D_MODEL, wp_sout + (size_t)i * D_MODEL * D_MODEL, D_MODEL,
            s_out_b + i * D_MODEL, nullptr, x, D_MODEL);

        // ---- GEGLU feed-forward block ----
        ln_rows<<<LN_GRID, 256, 0, stream>>>(x, f_ln_g + i * D_MODEL, f_ln_b + i * D_MODEL, xn, MPAD);
        gemm_bf16<3><<<gFF1, 256, 0, stream>>>(
            xn, D_MODEL, wp_fw1 + (size_t)i * D_MODEL * D_FF1, D_MODEL,
            f_b1 + i * D_FF1, nullptr, hbuf, D_FF1);
        geglu_k<<<cdiv((size_t)MPAD * D_FF2, 256), 256, 0, stream>>>(hbuf, ug);
        gemm_bf16<2><<<gOUT, 256, 0, stream>>>(
            ug, D_FF2, wp_fw2 + (size_t)i * D_FF2 * D_MODEL, D_FF2,
            f_b2 + i * D_MODEL, nullptr, x, D_MODEL);
    }

    head_k<<<1, 256, 0, stream>>>(x, o_ln_g, o_ln_b, o_w, o_b, out);
}
